// DepthsepCCBlock_1786706395754
// MI455X (gfx1250) — compile-verified, matrix operand order checked by reading, not run on hardware
//
#include <hip/hip_runtime.h>

typedef __bf16 bf16_t;
typedef __attribute__((ext_vector_type(16))) __bf16 v16bf;
typedef __attribute__((ext_vector_type(8)))  __bf16 v8bf;
typedef __attribute__((ext_vector_type(8)))  float   v8f;

constexpr int kN   = 2;
constexpr int kC   = 256;
constexpr int kH   = 128;
constexpr int kW   = 128;
constexpr int kHW  = kH * kW;        // 16384
constexpr int kHP  = kH + 2;         // padded (halo) height
constexpr int kWP  = kW + 2;         // padded (halo) width
constexpr int kSNC = 35;
constexpr int kNH  = 128;
constexpr int kHS  = 256;
constexpr int kK2  = kNH * 9;        // 1152 (implicit-GEMM K for 3x3 convs)
constexpr int kCS  = 136;            // LDS column stride (128 + 8 pad: lane m -> bank 4m)
constexpr float kEPS = 1e-5f;

#if __has_builtin(__builtin_amdgcn_sched_group_barrier)
// conv: per chunk [2 VMEM-read (A)] [8 DS-read (B)] [4 WMMA]
#define SCHED_CONV()                                      \
  do {                                                    \
    __builtin_amdgcn_sched_group_barrier(0x020, 2, 0);    \
    __builtin_amdgcn_sched_group_barrier(0x100, 8, 0);    \
    __builtin_amdgcn_sched_group_barrier(0x008, 4, 0);    \
  } while (0)
// gemm1x1: per chunk [10 VMEM-read] [4 WMMA]
#define SCHED_VMEM_THEN_WMMA()                            \
  do {                                                    \
    __builtin_amdgcn_sched_group_barrier(0x020, 10, 0);   \
    __builtin_amdgcn_sched_group_barrier(0x008, 4, 0);    \
  } while (0)
#else
#define SCHED_CONV() do {} while (0)
#define SCHED_VMEM_THEN_WMMA() do {} while (0)
#endif

// ---------------------------------------------------------------------------
// WMMA fragment helper (ISA 7.12.2: 16-bit A 16x32 / B 32x16 layout)
// lane L: row/col = L&15, khalf = (L>>4)*8
// fragment element e reads K = khalf + (e<8 ? e : 8+e)   (relative to chunk k0)
// => two contiguous 16-byte loads (global_load_b128 or ds_load_b128).
// ---------------------------------------------------------------------------
__device__ __forceinline__ v16bf load_frag16(const bf16_t* p) {
  v8bf lo = *(const v8bf*)(p);
  v8bf hi = *(const v8bf*)(p + 16);
  return __builtin_shufflevector(lo, hi, 0,1,2,3,4,5,6,7,8,9,10,11,12,13,14,15);
}

// ---------------------------------------------------------------------------
// Zero-fill (for the hid halo buffer). 16B per thread.
// ---------------------------------------------------------------------------
__global__ void zero_kernel(uint4* __restrict__ dst, long n16) {
  const long i = (long)blockIdx.x * blockDim.x + threadIdx.x;
  if (i < n16) dst[i] = make_uint4(0u, 0u, 0u, 0u);
}

// ---------------------------------------------------------------------------
// Kernel 1: four branch first convs (3x3, SNC->NH) fused with nearest
// downsample of seg (HS = 2H exactly -> segmap[y][x] = seg[2y][2x]).
// Output: hid[branch][n][y+1][x+1][cin]  (bf16, channels-last, 1-px zero halo)
// ---------------------------------------------------------------------------
__global__ void hid_kernel(const float* __restrict__ seg,
                           const float* w1a, const float* b1a,
                           const float* w1b, const float* b1b,
                           const float* w1c, const float* b1c,
                           const float* w1d, const float* b1d,
                           bf16_t* __restrict__ hid) {
  const int co = threadIdx.x;            // 0..127 output channel
  const int p  = blockIdx.x;             // pixel
  const int n  = blockIdx.y;
  const int br = blockIdx.z;             // branch 0..3
  const float* w1 = (br == 0) ? w1a : (br == 1) ? w1b : (br == 2) ? w1c : w1d;
  const float* b1 = (br == 0) ? b1a : (br == 1) ? b1b : (br == 2) ? b1c : b1d;
  const int y = p >> 7, x = p & 127;
  float acc = b1[co];
  const float* segn = seg + (size_t)n * kSNC * kHS * kHS;
#pragma unroll
  for (int tap = 0; tap < 9; ++tap) {
    const int yy = y + tap / 3 - 1;
    const int xx = x + tap % 3 - 1;
    if ((unsigned)yy < (unsigned)kH && (unsigned)xx < (unsigned)kW) {
      const float* sp = segn + (size_t)(2 * yy) * kHS + 2 * xx;
      const float* wp = w1 + ((size_t)co * kSNC) * 9 + tap;
      for (int ci = 0; ci < kSNC; ++ci)
        acc = fmaf(sp[(size_t)ci * kHS * kHS], wp[(size_t)ci * 9], acc);
    }
  }
  acc = acc > 0.f ? acc : 0.f;  // ReLU
  hid[(((size_t)(br * kN + n) * kHP + (y + 1)) * kWP + (x + 1)) * kNH + co] =
      (bf16_t)acc;
}

// ---------------------------------------------------------------------------
// Weight repack: (Cout, NH, 3, 3) f32 -> bf16 [Cout][k], k = tap*128 + cin
// ---------------------------------------------------------------------------
__global__ void pack_w2_kernel(const float* __restrict__ w2,
                               bf16_t* __restrict__ out, int cout_total) {
  const size_t i = (size_t)blockIdx.x * blockDim.x + threadIdx.x;
  if (i >= (size_t)cout_total * kK2) return;
  const int k   = (int)(i % kK2);
  const int co  = (int)(i / kK2);
  const int tap = k / kNH;
  const int ci  = k % kNH;
  out[i] = (bf16_t)w2[((size_t)co * kNH + ci) * 9 + tap];
}

__global__ void pack_w1x1_kernel(const float* __restrict__ w,
                                 bf16_t* __restrict__ out) {
  const int i = blockIdx.x * blockDim.x + threadIdx.x;  // (co,ci) flat, already k-major
  if (i < kC * kC) out[i] = (bf16_t)w[i];
}

// ---------------------------------------------------------------------------
// Kernel 3: implicit-GEMM 3x3 conv via WMMA, LDS-staged B panel.
// A block = 8 waves sharing one 64-pixel strip; wave w computes cout tile
// mt = mtg*8 + w. The whole im2col B panel for the strip (all 9 taps, K=1152)
// aliases into padded-hid rows y..y+2 x cols x0..x0+65 x 128ch = 50.7 KB,
// staged once into LDS (column stride 136 -> conflict-free ds_load_b128),
// then each wave runs 36 K-chunks: A from L2 (2 loads), B from LDS (8 ds),
// 4 WMMA chains. Global B traffic drops 8x vs fetching from L2 per wave.
// ---------------------------------------------------------------------------
__global__ void __launch_bounds__(256)
conv2_wmma_kernel(const bf16_t* __restrict__ Wt,    // [Cout][1152]
                  const bf16_t* __restrict__ hidp,  // [N][130][130][128]
                  const float*  __restrict__ bias,  // [Cout]
                  float*  __restrict__ sig_out,     // [N][Cout][HW] (sigmoid)
                  bf16_t* __restrict__ cw_out,      // [N][Cout][HW] (linear)
                  int Cout, int do_sigmoid) {
  __shared__ bf16_t sB[3 * 66 * kCS];               // 53.9 KB
  const int tid  = threadIdx.x;
  const int lane = tid & 31;
  const int widx = tid >> 5;                        // wave in block: 0..7
  const int colGroups = kN * kH * (kW / 64);        // 512
  const int cg  = blockIdx.x % colGroups;
  const int mtg = blockIdx.x / colGroups;
  const int n = cg >> 8;
  const int ct = cg & 255;
  const int y  = ct >> 1;
  const int x0 = (ct & 1) << 6;                     // 64-pixel strip
  const int mt = mtg * 8 + widx;

  // --- stage B window: padded rows y..y+2, cols x0..x0+65, 128 ch each ---
  {
    const bf16_t* src = hidp + (size_t)n * kHP * kWP * kNH
                      + ((size_t)y * kWP + x0) * kNH;
    // 198 columns x 16 v8bf-chunks each
    for (int idx = tid; idx < 198 * 16; idx += 256) {
      const int colIdx = idx >> 4;        // 0..197
      const int part   = idx & 15;        // 16B chunk within the 128-ch column
      const int r = colIdx / 66;
      const int c = colIdx % 66;
      *(v8bf*)(sB + (size_t)colIdx * kCS + part * 8) =
          *(const v8bf*)(src + ((size_t)r * kWP + c) * kNH + part * 8);
    }
  }
  __syncthreads();

  const int mrow = lane & 15;
  const int kh   = (lane >> 4) << 3;                // 0 or 8
  const bf16_t* aRow = Wt + (size_t)(mt * 16 + mrow) * kK2 + kh;
  const bf16_t* bL   = sB + (size_t)mrow * kCS + kh;  // lane base into LDS window

  // chunk ch (0..35): tap = ch>>2, c0 = (ch&3)*32
  #define AOFF(ch) ((size_t)(((ch) >> 2) * kNH + (((ch) & 3) << 5)))
  #define BO(ch, j) ((size_t)((((((ch) >> 2) / 3) * 66) + (((ch) >> 2) % 3) + (j) * 16)) * kCS + (((ch) & 3) << 5))

  v8f acc0 = {};
  v8f acc1 = {};
  v8f acc2 = {};
  v8f acc3 = {};
#pragma unroll
  for (int ch = 0; ch < 36; ++ch) {
    v16bf a  = load_frag16(aRow + AOFF(ch));
    v16bf b0 = load_frag16(bL + BO(ch, 0));
    v16bf b1 = load_frag16(bL + BO(ch, 1));
    v16bf b2 = load_frag16(bL + BO(ch, 2));
    v16bf b3 = load_frag16(bL + BO(ch, 3));
    acc0 = __builtin_amdgcn_wmma_f32_16x16x32_bf16(false, a, false, b0,
                                                   (short)0, acc0, false, false);
    acc1 = __builtin_amdgcn_wmma_f32_16x16x32_bf16(false, a, false, b1,
                                                   (short)0, acc1, false, false);
    acc2 = __builtin_amdgcn_wmma_f32_16x16x32_bf16(false, a, false, b2,
                                                   (short)0, acc2, false, false);
    acc3 = __builtin_amdgcn_wmma_f32_16x16x32_bf16(false, a, false, b3,
                                                   (short)0, acc3, false, false);
    SCHED_CONV();
  }
  #undef AOFF
  #undef BO

  // C/D layout: lane L -> col = L&15, rows = r + (L>=16 ? 8 : 0)
  const int col = lane & 15;
  const int rb  = (lane >> 4) << 3;
#pragma unroll
  for (int r = 0; r < 8; ++r) {
    const int co = mt * 16 + rb + r;
    const float bv = bias[co];
    const size_t base = ((size_t)n * Cout + co) * kHW + (size_t)y * kW + x0 + col;
    const float v0 = acc0[r] + bv;
    const float v1 = acc1[r] + bv;
    const float v2 = acc2[r] + bv;
    const float v3 = acc3[r] + bv;
    if (do_sigmoid) {
      sig_out[base]      = 1.f / (1.f + __expf(-v0));
      sig_out[base + 16] = 1.f / (1.f + __expf(-v1));
      sig_out[base + 32] = 1.f / (1.f + __expf(-v2));
      sig_out[base + 48] = 1.f / (1.f + __expf(-v3));
    } else {
      cw_out[base]      = (bf16_t)v0;
      cw_out[base + 16] = (bf16_t)v1;
      cw_out[base + 32] = (bf16_t)v2;
      cw_out[base + 48] = (bf16_t)v3;
    }
  }
}

// ---------------------------------------------------------------------------
// Kernel 4: training-mode BN stats -> per-channel scale/shift
//   scale = g * rsqrt(var+eps), shift = b - mean*scale
// ---------------------------------------------------------------------------
__global__ void bn_stats_kernel(const float* __restrict__ src,
                                const float* __restrict__ g,
                                const float* __restrict__ b,
                                float* __restrict__ scale,
                                float* __restrict__ shift) {
  const int c = blockIdx.x;
  float s = 0.f, s2 = 0.f;
  for (int n = 0; n < kN; ++n) {
    const float* p = src + ((size_t)n * kC + c) * kHW;
    for (int i = threadIdx.x; i < kHW; i += blockDim.x) {
      const float v = p[i];
      s += v;
      s2 = fmaf(v, v, s2);
    }
  }
  __shared__ float rs[256], rs2[256];
  rs[threadIdx.x] = s; rs2[threadIdx.x] = s2;
  __syncthreads();
  for (int o = 128; o > 0; o >>= 1) {
    if (threadIdx.x < o) {
      rs[threadIdx.x]  += rs[threadIdx.x + o];
      rs2[threadIdx.x] += rs2[threadIdx.x + o];
    }
    __syncthreads();
  }
  if (threadIdx.x == 0) {
    const float inv  = 1.f / (float)(kN * kHW);
    const float mean = rs[0] * inv;
    const float var  = rs2[0] * inv - mean * mean;
    const float sc   = g[c] * rsqrtf(var + kEPS);
    scale[c] = sc;
    shift[c] = b[c] - mean * sc;
  }
}

// ---------------------------------------------------------------------------
// Kernel 5: per-pixel dynamic depthwise 3x3 conv, BN folded into reads.
//   t[n][p][c] = sum_tap cw[n][c*9+tap][p] * (scale[c]*src[n][c][nbr(p,tap)]+shift[c])
// (zero padding after BN == skip tap, matches reference)
// ---------------------------------------------------------------------------
__global__ void depthconv_kernel(const float*  __restrict__ src,    // [N][C][HW]
                                 const float*  __restrict__ scale,
                                 const float*  __restrict__ shift,
                                 const bf16_t* __restrict__ cw,     // [N][C*9][HW]
                                 bf16_t* __restrict__ t) {          // [N][HW][C]
  const int c = threadIdx.x;   // 256
  const int p = blockIdx.x;    // HW
  const int n = blockIdx.y;
  const int y = p >> 7, x = p & 127;
  const float sc = scale[c], sh = shift[c];
  const float*  sp  = src + ((size_t)n * kC + c) * kHW;
  const bf16_t* cwp = cw + ((size_t)n * kC * 9 + (size_t)c * 9) * kHW + p;
  float acc = 0.f;
#pragma unroll
  for (int tap = 0; tap < 9; ++tap) {
    const int yy = y + tap / 3 - 1;
    const int xx = x + tap % 3 - 1;
    if ((unsigned)yy < (unsigned)kH && (unsigned)xx < (unsigned)kW) {
      const float v = fmaf(sp[(size_t)yy * kW + xx], sc, sh);
      acc = fmaf((float)cwp[(size_t)tap * kHW], v, acc);
    }
  }
  t[((size_t)n * kHW + p) * kC + c] = (bf16_t)acc;
}

// ---------------------------------------------------------------------------
// Kernel 6: 1x1 conv as WMMA GEMM (256x256 @ K=256), 16 couts x 64 pixels per
// wave, pinned [VMEM][WMMA] groups, fused epilogue:
//   v = (acc + bias) * sw ; v = lrelu(v) ; [+ residual x] -> dst (f32 NCHW)
// ---------------------------------------------------------------------------
__global__ void gemm1x1_wmma_kernel(const bf16_t* __restrict__ Wp,    // [256][256]
                                    const bf16_t* __restrict__ t,     // [N][HW][256]
                                    const float*  __restrict__ bias,  // [256]
                                    const float*  __restrict__ sw,    // [N][256][HW]
                                    const float*  __restrict__ resid, // x or nullptr
                                    float* __restrict__ dst) {        // [N][256][HW]
  const int lane = threadIdx.x & 31;
  const int wave = blockIdx.x * (blockDim.x >> 5) + (threadIdx.x >> 5);
  const int Mt = kC >> 4;                    // 16
  const int colGroups = kN * (kHW >> 6);     // 512
  if (wave >= Mt * colGroups) return;
  const int mt = wave / colGroups;
  int ct = wave - mt * colGroups;
  const int n  = ct >> 8;
  const int p0 = (ct & 255) << 6;            // 64-pixel strip
  const int mrow = lane & 15;
  const int kh   = (lane >> 4) << 3;
  const bf16_t* aRow = Wp + (size_t)(mt * 16 + mrow) * kC + kh;
  const bf16_t* bRow = t + ((size_t)n * kHW + p0 + mrow) * kC + kh;
  v8f acc0 = {};
  v8f acc1 = {};
  v8f acc2 = {};
  v8f acc3 = {};
#pragma unroll
  for (int ch = 0; ch < 8; ++ch) {
    const int k0 = ch * 32;
    v16bf a  = load_frag16(aRow + k0);
    v16bf b0 = load_frag16(bRow + k0);
    v16bf b1 = load_frag16(bRow + 16 * kC + k0);
    v16bf b2 = load_frag16(bRow + 32 * kC + k0);
    v16bf b3 = load_frag16(bRow + 48 * kC + k0);
    acc0 = __builtin_amdgcn_wmma_f32_16x16x32_bf16(false, a, false, b0,
                                                   (short)0, acc0, false, false);
    acc1 = __builtin_amdgcn_wmma_f32_16x16x32_bf16(false, a, false, b1,
                                                   (short)0, acc1, false, false);
    acc2 = __builtin_amdgcn_wmma_f32_16x16x32_bf16(false, a, false, b2,
                                                   (short)0, acc2, false, false);
    acc3 = __builtin_amdgcn_wmma_f32_16x16x32_bf16(false, a, false, b3,
                                                   (short)0, acc3, false, false);
    SCHED_VMEM_THEN_WMMA();
  }
  const int col = lane & 15;
  const int rb  = (lane >> 4) << 3;
#pragma unroll
  for (int r = 0; r < 8; ++r) {
    const int co = mt * 16 + rb + r;
    const float bv = bias[co];
    const size_t idx = ((size_t)n * kC + co) * kHW + p0 + col;
    float v[4] = {acc0[r] + bv, acc1[r] + bv, acc2[r] + bv, acc3[r] + bv};
#pragma unroll
    for (int j = 0; j < 4; ++j) {
      float u = v[j] * sw[idx + j * 16];
      u = u >= 0.f ? u : 0.2f * u;           // LeakyReLU(0.2)
      if (resid) u += resid[idx + j * 16];
      dst[idx + j * 16] = u;
    }
  }
}

// ---------------------------------------------------------------------------
extern "C" void kernel_launch(void* const* d_in, const int* in_sizes, int n_in,
                              void* d_out, int out_size, void* d_ws, size_t ws_size,
                              hipStream_t stream) {
  (void)in_sizes; (void)n_in; (void)out_size; (void)ws_size;
  const float* x       = (const float*)d_in[0];
  const float* seg     = (const float*)d_in[1];
  const float* gw1_w1  = (const float*)d_in[2];
  const float* gw1_b1  = (const float*)d_in[3];
  const float* gw1_w2  = (const float*)d_in[4];
  const float* gw1_b2  = (const float*)d_in[5];
  const float* gw2_w1  = (const float*)d_in[6];
  const float* gw2_b1  = (const float*)d_in[7];
  const float* gw2_w2  = (const float*)d_in[8];
  const float* gw2_b2  = (const float*)d_in[9];
  const float* se1_w1  = (const float*)d_in[10];
  const float* se1_b1  = (const float*)d_in[11];
  const float* se1_w2  = (const float*)d_in[12];
  const float* se1_b2  = (const float*)d_in[13];
  const float* se2_w1  = (const float*)d_in[14];
  const float* se2_b1  = (const float*)d_in[15];
  const float* se2_w2  = (const float*)d_in[16];
  const float* se2_b2  = (const float*)d_in[17];
  const float* conv1_w = (const float*)d_in[18];
  const float* conv1_b = (const float*)d_in[19];
  const float* conv3_w = (const float*)d_in[20];
  const float* conv3_b = (const float*)d_in[21];
  const float* bn1_g   = (const float*)d_in[22];
  const float* bn1_b   = (const float*)d_in[23];
  const float* bn2_g   = (const float*)d_in[24];
  const float* bn2_b   = (const float*)d_in[25];

  // scratch layout
  char* wsb = (char*)d_ws;
  size_t off = 0;
  auto take = [&](size_t bytes) -> void* {
    void* p = wsb + off;
    off += (bytes + 255) & ~(size_t)255;
    return p;
  };
  const size_t hidBranchElems = (size_t)kN * kHP * kWP * kNH;  // padded
  const size_t hidBytes = 4 * hidBranchElems * sizeof(bf16_t); // 34.6 MB
  bf16_t* hid  = (bf16_t*)take(hidBytes);
  bf16_t* wpg1 = (bf16_t*)take((size_t)(kC * 9) * kK2 * sizeof(bf16_t));
  bf16_t* wpg2 = (bf16_t*)take((size_t)(kC * 9) * kK2 * sizeof(bf16_t));
  bf16_t* wps1 = (bf16_t*)take((size_t)kC * kK2 * sizeof(bf16_t));
  bf16_t* wps2 = (bf16_t*)take((size_t)kC * kK2 * sizeof(bf16_t));
  bf16_t* wpc1 = (bf16_t*)take((size_t)kC * kC * sizeof(bf16_t));
  bf16_t* wpc3 = (bf16_t*)take((size_t)kC * kC * sizeof(bf16_t));
  bf16_t* cw1  = (bf16_t*)take((size_t)kN * kC * 9 * kHW * sizeof(bf16_t)); // 151 MB
  bf16_t* cw2  = (bf16_t*)take((size_t)kN * kC * 9 * kHW * sizeof(bf16_t)); // 151 MB
  float*  sw1  = (float*)take((size_t)kN * kC * kHW * sizeof(float));       // 33.6 MB
  float*  sw2  = (float*)take((size_t)kN * kC * kHW * sizeof(float));
  bf16_t* tbuf = (bf16_t*)take((size_t)kN * kHW * kC * sizeof(bf16_t));     // 16.8 MB
  float*  dx2  = (float*)take((size_t)kN * kC * kHW * sizeof(float));       // 33.6 MB
  float*  ss   = (float*)take(4 * kC * sizeof(float));  // scale1,shift1,scale2,shift2

  // 0) zero the padded hid buffer (halo must be zero for unpredicated loads)
  {
    const long n16 = (long)(hidBytes / 16);
    zero_kernel<<<(int)((n16 + 255) / 256), 256, 0, stream>>>((uint4*)hid, n16);
  }

  // 1) branch first convs (+nearest downsample, ReLU) -> bf16 padded NHWC
  hid_kernel<<<dim3(kHW, kN, 4), kNH, 0, stream>>>(
      seg, gw1_w1, gw1_b1, gw2_w1, gw2_b1, se1_w1, se1_b1, se2_w1, se2_b1, hid);

  // 2) weight repacks
  {
    const size_t tg = (size_t)(kC * 9) * kK2;
    pack_w2_kernel<<<(int)((tg + 255) / 256), 256, 0, stream>>>(gw1_w2, wpg1, kC * 9);
    pack_w2_kernel<<<(int)((tg + 255) / 256), 256, 0, stream>>>(gw2_w2, wpg2, kC * 9);
    const size_t ts = (size_t)kC * kK2;
    pack_w2_kernel<<<(int)((ts + 255) / 256), 256, 0, stream>>>(se1_w2, wps1, kC);
    pack_w2_kernel<<<(int)((ts + 255) / 256), 256, 0, stream>>>(se2_w2, wps2, kC);
    pack_w1x1_kernel<<<(kC * kC + 255) / 256, 256, 0, stream>>>(conv1_w, wpc1);
    pack_w1x1_kernel<<<(kC * kC + 255) / 256, 256, 0, stream>>>(conv3_w, wpc3);
  }

  // 3) WMMA implicit-GEMM 3x3 convs: blocks = (Cout/128) * 512 strips
  auto c2blocks = [](int Cout) { return (Cout / 128) * (kN * kH * (kW / 64)); };
  conv2_wmma_kernel<<<c2blocks(kC * 9), 256, 0, stream>>>(
      wpg1, hid + 0 * hidBranchElems, gw1_b2, nullptr, cw1, kC * 9, 0);
  conv2_wmma_kernel<<<c2blocks(kC * 9), 256, 0, stream>>>(
      wpg2, hid + 1 * hidBranchElems, gw2_b2, nullptr, cw2, kC * 9, 0);
  conv2_wmma_kernel<<<c2blocks(kC), 256, 0, stream>>>(
      wps1, hid + 2 * hidBranchElems, se1_b2, sw1, nullptr, kC, 1);
  conv2_wmma_kernel<<<c2blocks(kC), 256, 0, stream>>>(
      wps2, hid + 3 * hidBranchElems, se2_b2, sw2, nullptr, kC, 1);

  // 4) main path, stage 1: BN1 -> depthconv(cw1) -> 1x1 conv -> *sw1 -> lrelu
  bn_stats_kernel<<<kC, 256, 0, stream>>>(x, bn1_g, bn1_b, ss, ss + kC);
  depthconv_kernel<<<dim3(kHW, kN), kC, 0, stream>>>(x, ss, ss + kC, cw1, tbuf);
  const int g1blocks = ((kC >> 4) * (kN * (kHW >> 6))) / 8;  // 1024
  gemm1x1_wmma_kernel<<<g1blocks, 256, 0, stream>>>(
      wpc1, tbuf, conv1_b, sw1, nullptr, dx2);

  // 5) stage 2: BN2 -> depthconv(cw2) -> 1x1 conv -> *sw2 -> lrelu -> +x
  bn_stats_kernel<<<kC, 256, 0, stream>>>(dx2, bn2_g, bn2_b, ss + 2 * kC, ss + 3 * kC);
  depthconv_kernel<<<dim3(kHW, kN), kC, 0, stream>>>(dx2, ss + 2 * kC, ss + 3 * kC, cw2, tbuf);
  gemm1x1_wmma_kernel<<<g1blocks, 256, 0, stream>>>(
      wpc3, tbuf, conv3_b, sw2, x, (float*)d_out);
}
